// PLELayer_6511170420803
// MI455X (gfx1250) — matrix operand order repaired and checked by reading, not compiled, not run on hardware
//
#include <hip/hip_runtime.h>
#include <stdint.h>

// ---------------- problem constants ----------------
#define TT 4
#define BB 8192
#define DD 1024
#define UU 512
#define EE 8
#define NN 4096   // U*E
#define GN 16     // 2*E

typedef __attribute__((ext_vector_type(16))) __bf16 v16bf;
typedef __attribute__((ext_vector_type(8)))  float  v8f;

static __device__ __forceinline__ unsigned short f2bf(float f) {
  unsigned u = __builtin_bit_cast(unsigned, f);
  unsigned r = u + 0x7FFFu + ((u >> 16) & 1u);   // round-to-nearest-even
  return (unsigned short)(r >> 16);
}
static __device__ __forceinline__ float bf2f(unsigned short h) {
  return __builtin_bit_cast(float, ((unsigned)h) << 16);
}
static __device__ __forceinline__ unsigned packbf(float lo, float hi) {
  return (unsigned)f2bf(lo) | ((unsigned)f2bf(hi) << 16);
}

union Frag { uint4 q[2]; v16bf v; };
union HQ   { uint4 q; unsigned short h[8]; };

// Async global->LDS copy of 32 contiguous bytes (2 x B128), per-lane addresses.
static __device__ __forceinline__ void async_copy32(unsigned ldsAddr, const void* gaddr) {
  uint64_t ga = (uint64_t)(uintptr_t)gaddr;
  asm volatile("global_load_async_to_lds_b128 %0, %1, off"
               :: "v"(ldsAddr), "v"(ga) : "memory");
  asm volatile("global_load_async_to_lds_b128 %0, %1, off offset:16"
               :: "v"(ldsAddr), "v"(ga) : "memory");
}
static __device__ __forceinline__ void wait_async0() {
  asm volatile("s_wait_asynccnt 0x0" ::: "memory");
}

// ---------------- main expert GEMM ----------------
// C(128x128) tile per block; 8 waves as 4(M) x 2(N); each wave: 2x4 wmma tiles.
// Double-buffered async global->LDS staging: tile k+1 DMA overlaps tile k WMMAs.
#define BM 128
#define BN 128
#define BK 32
#define LDS_STR 20   // dwords per LDS row: conflict-free ds_load_b128, 16B aligned

union Smem {
  struct { unsigned a[2][BM * LDS_STR]; unsigned b[2][BN * LDS_STR]; } st; // 41 KB
  unsigned short et[BM * BN];                                              // 32 KB
};

// EPI: 0 = level0 shared experts (store es bf16 + share partial, gate off 32)
//      1 = level0 task experts   (combine with es/g0 -> task_out bf16 + share partial t*8)
//      2 = level1 shared experts (store es1 bf16 only)
//      3 = level1 task experts   (combine with es1/g1 -> fp32 final output)
template <int EPI>
__global__ __launch_bounds__(256) void gemm_ple(
    const unsigned short* __restrict__ A, size_t aTStride,
    const unsigned short* __restrict__ Wbf,      // bf16, transposed [n][k], slices of NN*K
    const float* __restrict__ bias, const float* __restrict__ tGate,
    const float* __restrict__ sGate, float* __restrict__ sPart,
    const unsigned short* __restrict__ esIn, unsigned short* __restrict__ bfOut,
    float* __restrict__ fOut, int lda, int K) {
  __shared__ Smem sm;
  const int tid = threadIdx.x;
  const int t   = blockIdx.z;
  const int mg0 = blockIdx.y * BM;
  const int ng0 = blockIdx.x * BN;
  const unsigned short* At = A + aTStride * t;
  const unsigned short* Wt = Wbf + (size_t)t * NN * K;
  const float* bt = bias + (size_t)t * NN;

  const int lane = tid & 31, wave = tid >> 5;
  const int wy = wave >> 1, wx = wave & 1;       // 4x2 wave grid
  const int r = lane >> 4, l15 = lane & 15;

  v8f acc[2][4];
#pragma unroll
  for (int i = 0; i < 2; i++)
#pragma unroll
    for (int j = 0; j < 4; j++)
#pragma unroll
      for (int p = 0; p < 8; p++) acc[i][j][p] = 0.f;

  // staging mapping: 256 threads = 128 rows x 2 halves (16 bf16 each)
  const int srow = tid >> 1, shalf = tid & 1;
  const unsigned short* aSrc = At + (size_t)(mg0 + srow) * lda + shalf * 16;
  const unsigned short* bSrc = Wt + (size_t)(ng0 + srow) * K  + shalf * 16;
  unsigned ldsA[2], ldsB[2];
#pragma unroll
  for (int bfi = 0; bfi < 2; bfi++) {
    ldsA[bfi] = (unsigned)(uintptr_t)&sm.st.a[bfi][srow * LDS_STR + shalf * 8];
    ldsB[bfi] = (unsigned)(uintptr_t)&sm.st.b[bfi][srow * LDS_STR + shalf * 8];
  }

  // prologue: stage tile 0 into buffer 0
  async_copy32(ldsA[0], aSrc);
  async_copy32(ldsB[0], bSrc);
  wait_async0();
  __syncthreads();

  int cur = 0;
  for (int k0 = 0; k0 < K; k0 += BK) {
    // issue DMA for next tile into the other buffer (overlaps with WMMAs below)
    if (k0 + BK < K) {
      async_copy32(ldsA[cur ^ 1], aSrc + k0 + BK);
      async_copy32(ldsB[cur ^ 1], bSrc + k0 + BK);
      if (k0 + 2 * BK < K) {
        __builtin_prefetch(aSrc + k0 + 2 * BK, 0, 3);
        __builtin_prefetch(bSrc + k0 + 2 * BK, 0, 3);
      }
    }
    // ---- fragments per ISA layout + 8 WMMAs on current buffer ----
    const unsigned* abuf = sm.st.a[cur];
    const unsigned* bbuf = sm.st.b[cur];
    Frag fa[2], fb[4];
#pragma unroll
    for (int i = 0; i < 2; i++) {
      const unsigned* arp = &abuf[(wy * 32 + i * 16 + l15) * LDS_STR];
      fa[i].q[0] = *(const uint4*)(arp + 4 * r);       // K 0..7  (r=0) / 8..15
      fa[i].q[1] = *(const uint4*)(arp + 8 + 4 * r);   // K 16..23 / 24..31
    }
#pragma unroll
    for (int j = 0; j < 4; j++) {
      const unsigned* brp = &bbuf[(wx * 64 + j * 16 + l15) * LDS_STR];
      fb[j].q[0] = *(const uint4*)(brp + 4 * r);       // K 0..7  (r=0) / 8..15
      fb[j].q[1] = *(const uint4*)(brp + 8 + 4 * r);   // K 16..23 / 24..31
    }
#pragma unroll
    for (int i = 0; i < 2; i++)
#pragma unroll
      for (int j = 0; j < 4; j++)
        acc[i][j] = __builtin_amdgcn_wmma_f32_16x16x32_bf16(
            false, fa[i].v, false, fb[j].v, (short)0, acc[i][j], false, false);
    // next tile complete + all waves done reading current buffer
    wait_async0();
    __syncthreads();
    cur ^= 1;
  }

  // ---------------- epilogue ----------------
  float bj[4];
#pragma unroll
  for (int j = 0; j < 4; j++) bj[j] = bt[ng0 + wx * 64 + j * 16 + l15];

  const int sgOff = (EPI == 0) ? 32 : t * EE;
  float* spT = sPart ? (sPart + (size_t)(EPI == 0 ? 0 : t) * (size_t)BB * UU) : nullptr;

#pragma unroll
  for (int i = 0; i < 2; i++) {
#pragma unroll
    for (int j = 0; j < 4; j++) {
#pragma unroll
      for (int p = 0; p < 8; p++) {
        const int ml = wy * 32 + i * 16 + p + 8 * r;   // C layout: VGPR p -> M=p / p+8
        const int nl = wx * 64 + j * 16 + l15;
        const int mg = mg0 + ml, ng = ng0 + nl;
        float v = acc[i][j][p] + bj[j];
        v = v > 0.f ? v : 0.f;                          // ReLU
        if (EPI == 0 || EPI == 2)
          bfOut[(size_t)mg * NN + ng] = f2bf(v);        // es / es1
        if (EPI == 1 || EPI == 3)
          sm.et[ml * BN + nl] = f2bf(v);                // stage et for combine
        if (EPI == 0 || EPI == 1) {                     // level-0 share-gate partial
          float s = v * sGate[(size_t)mg * 40 + sgOff + (lane & 7)];
          s += __shfl_xor(s, 1);
          s += __shfl_xor(s, 2);
          s += __shfl_xor(s, 4);
          if ((lane & 7) == 0) spT[(size_t)mg * UU + (ng >> 3)] = s;
        }
      }
    }
  }

  if (EPI == 1 || EPI == 3) {                           // task combine over 8 experts
    __syncthreads();
    const int u = tid & 15, m0 = tid >> 4;
#pragma unroll
    for (int jj = 0; jj < 8; jj++) {
      const int ml = m0 + jj * 16;
      const int mg = mg0 + ml;
      HQ et; et.q = *(const uint4*)&sm.et[ml * BN + u * 8];
      HQ es; es.q = *(const uint4*)&esIn[(size_t)mg * NN + ng0 + u * 8];
      const float* gv = tGate + ((size_t)t * BB + mg) * GN;
      float o = 0.f;
#pragma unroll
      for (int e = 0; e < 8; e++)
        o += bf2f(et.h[e]) * gv[e] + bf2f(es.h[e]) * gv[8 + e];
      const int ug = (ng0 >> 3) + u;
      if (EPI == 1) bfOut[((size_t)t * BB + mg) * UU + ug] = f2bf(o);
      else          fOut [((size_t)t * BB + mg) * UU + ug] = o;
    }
  }
}

// ---------------- one-time fp32 -> bf16 converts ----------------
__global__ void cvt_x_kernel(const float* __restrict__ src, unsigned* __restrict__ dst) {
  size_t i = (size_t)blockIdx.x * 256 + threadIdx.x;   // dword index
  dst[i] = packbf(src[2 * i], src[2 * i + 1]);
}

// W (K,NN) fp32 -> Wt (NN,K) bf16, slice blockIdx.z; LDS-tiled 32(k) x 128(n).
__global__ __launch_bounds__(256) void cvt_wT_kernel(
    const float* __restrict__ src, unsigned short* __restrict__ dst, int K) {
  __shared__ float tile[128][33];
  const int tid = threadIdx.x;
  const int n0 = blockIdx.x * 128;
  const int k0 = blockIdx.y * 32;
  const float* sp = src + (size_t)blockIdx.z * K * NN;
  unsigned short* dp = dst + (size_t)blockIdx.z * NN * K;
#pragma unroll
  for (int it = 0; it < 16; it++) {
    int flat = it * 256 + tid;                // 4096 elements
    int k = flat >> 7, n = flat & 127;        // coalesced along n
    tile[n][k] = sp[(size_t)(k0 + k) * NN + n0 + n];
  }
  __syncthreads();
#pragma unroll
  for (int it = 0; it < 8; it++) {
    int flat = it * 256 + tid;                // 2048 dwords
    int n = flat >> 4, kd = flat & 15;        // coalesced along k
    unsigned v = packbf(tile[n][2 * kd], tile[n][2 * kd + 1]);
    *(unsigned*)&dp[(size_t)(n0 + n) * K + k0 + 2 * kd] = v;
  }
}

// ---------------- level-0 gates: g0 (T,B,16) softmax16, gs (B,40) softmax40 ----------------
__global__ __launch_bounds__(128) void gates0_kernel(
    const float* __restrict__ x, const float* __restrict__ gw0,
    const float* __restrict__ gb0, const float* __restrict__ gws,
    const float* __restrict__ gbs, float* __restrict__ g0, float* __restrict__ gs) {
  __shared__ float xs[8 * DD];
  __shared__ float zb[8][104];
  const int tid = threadIdx.x;
  const int b0 = blockIdx.x * 8;
  const float4* src = (const float4*)(x + (size_t)b0 * DD);
  float4* dst = (float4*)xs;
#pragma unroll
  for (int it = 0; it < 16; it++) dst[it * 128 + tid] = src[it * 128 + tid];
  __syncthreads();
  if (tid < 104) {
    const float* wp; int stride; float bv;
    if (tid < 64) { int tt = tid >> 4, c = tid & 15;
      wp = gw0 + (size_t)tt * DD * GN + c; stride = GN; bv = gb0[tt * GN + c];
    } else { int c = tid - 64; wp = gws + c; stride = 40; bv = gbs[c]; }
    float accv[8] = {0, 0, 0, 0, 0, 0, 0, 0};
    for (int d = 0; d < DD; d++) {
      float w = wp[(size_t)d * stride];
#pragma unroll
      for (int rr = 0; rr < 8; rr++) accv[rr] += xs[rr * DD + d] * w;
    }
    for (int rr = 0; rr < 8; rr++) zb[rr][tid] = accv[rr] + bv;
  }
  __syncthreads();
  if (tid < 40) {
    const int rr = tid / 5, grp = tid % 5;
    const int base = (grp < 4) ? grp * 16 : 64;
    const int n = (grp < 4) ? 16 : 40;
    float m = -1e30f;
    for (int k = 0; k < n; k++) m = fmaxf(m, zb[rr][base + k]);
    float ssum = 0.f;
    for (int k = 0; k < n; k++) ssum += __expf(zb[rr][base + k] - m);
    float inv = 1.f / ssum;
    if (grp < 4)
      for (int k = 0; k < GN; k++)
        g0[((size_t)grp * BB + b0 + rr) * GN + k] = __expf(zb[rr][base + k] - m) * inv;
    else
      for (int k = 0; k < 40; k++)
        gs[(size_t)(b0 + rr) * 40 + k] = __expf(zb[rr][base + k] - m) * inv;
  }
}

// ---------------- level-1 gates from bf16 task_out ----------------
__global__ __launch_bounds__(128) void gates1_kernel(
    const unsigned short* __restrict__ tob, const float* __restrict__ gw1,
    const float* __restrict__ gb1, float* __restrict__ g1) {
  __shared__ float xs[8 * UU];
  __shared__ float zb[8][GN];
  const int tid = threadIdx.x;
  const int b0 = blockIdx.x * 8;
  const int t = blockIdx.y;
  const unsigned* src = (const unsigned*)(tob + ((size_t)t * BB + b0) * UU);
#pragma unroll
  for (int it = 0; it < 16; it++) {
    unsigned d = src[it * 128 + tid];
    int idx = (it * 128 + tid) * 2;
    xs[idx] = bf2f((unsigned short)(d & 0xFFFFu));
    xs[idx + 1] = bf2f((unsigned short)(d >> 16));
  }
  __syncthreads();
  if (tid < GN) {
    const float* wp = gw1 + (size_t)t * UU * GN + tid;
    const float bv = gb1[t * GN + tid];
    float accv[8] = {0, 0, 0, 0, 0, 0, 0, 0};
    for (int d = 0; d < UU; d++) {
      float w = wp[(size_t)d * GN];
#pragma unroll
      for (int rr = 0; rr < 8; rr++) accv[rr] += xs[rr * UU + d] * w;
    }
    for (int rr = 0; rr < 8; rr++) zb[rr][tid] = accv[rr] + bv;
  }
  __syncthreads();
  if (tid < 8) {
    float m = -1e30f;
    for (int k = 0; k < GN; k++) m = fmaxf(m, zb[tid][k]);
    float ssum = 0.f;
    for (int k = 0; k < GN; k++) ssum += __expf(zb[tid][k] - m);
    float inv = 1.f / ssum;
    for (int k = 0; k < GN; k++)
      g1[((size_t)t * BB + b0 + tid) * GN + k] = __expf(zb[tid][k] - m) * inv;
  }
}

// ---------------- sum 5 share partials -> bf16 share input ----------------
__global__ void share_cvt_kernel(const float* __restrict__ sp,
                                 unsigned short* __restrict__ outb) {
  const size_t BU = (size_t)BB * UU;
  size_t i = (size_t)blockIdx.x * 256 + threadIdx.x;
  float s = sp[i] + sp[BU + i] + sp[2 * BU + i] + sp[3 * BU + i] + sp[4 * BU + i];
  outb[i] = f2bf(s);
}

// ---------------- host orchestration ----------------
extern "C" void kernel_launch(void* const* d_in, const int* in_sizes, int n_in,
                              void* d_out, int out_size, void* d_ws, size_t ws_size,
                              hipStream_t stream) {
  const float* x    = (const float*)d_in[0];
  const float* ews0 = (const float*)d_in[1];
  const float* ebs0 = (const float*)d_in[2];
  const float* ews1 = (const float*)d_in[3];
  const float* ebs1 = (const float*)d_in[4];
  const float* gws  = (const float*)d_in[5];
  const float* gbs  = (const float*)d_in[6];
  const float* ew0  = (const float*)d_in[7];
  const float* eb0  = (const float*)d_in[8];
  const float* gw0  = (const float*)d_in[9];
  const float* gb0  = (const float*)d_in[10];
  const float* ew1  = (const float*)d_in[11];
  const float* eb1  = (const float*)d_in[12];
  const float* gw1  = (const float*)d_in[13];
  const float* gb1  = (const float*)d_in[14];

  char* ws = (char*)d_ws;
  const size_t BU = (size_t)BB * UU;
  size_t off = 0;
  unsigned short* xb    = (unsigned short*)(ws + off); off += (size_t)BB * DD * 2;
  unsigned short* w0sT  = (unsigned short*)(ws + off); off += (size_t)NN * DD * 2;
  unsigned short* w0tT  = (unsigned short*)(ws + off); off += (size_t)TT * NN * DD * 2;
  unsigned short* w1sT  = (unsigned short*)(ws + off); off += (size_t)NN * UU * 2;
  unsigned short* w1tT  = (unsigned short*)(ws + off); off += (size_t)TT * NN * UU * 2;
  float*          sPart = (float*)(ws + off);          off += 5 * BU * sizeof(float);
  unsigned short* shareBf = (unsigned short*)(ws + off); off += BU * 2;
  unsigned short* esBf  = (unsigned short*)(ws + off); off += (size_t)BB * NN * 2;
  unsigned short* toBf  = (unsigned short*)(ws + off); off += (size_t)TT * BU * 2;
  float*          g0    = (float*)(ws + off); off += (size_t)TT * BB * GN * sizeof(float);
  float*          gsArr = (float*)(ws + off); off += (size_t)BB * 40 * sizeof(float);
  float*          g1    = (float*)(ws + off);

  const dim3 blk(256);
  const dim3 g1z(NN / BN, BB / BM, 1);
  const dim3 g4z(NN / BN, BB / BM, TT);

  // one-time conversions (bf16, weights transposed to [n][k])
  cvt_x_kernel<<<(BB * DD / 2) / 256, 256, 0, stream>>>(x, (unsigned*)xb);
  cvt_wT_kernel<<<dim3(NN / 128, DD / 32, 1), blk, 0, stream>>>(ews0, w0sT, DD);
  cvt_wT_kernel<<<dim3(NN / 128, DD / 32, TT), blk, 0, stream>>>(ew0, w0tT, DD);
  cvt_wT_kernel<<<dim3(NN / 128, UU / 32, 1), blk, 0, stream>>>(ews1, w1sT, UU);
  cvt_wT_kernel<<<dim3(NN / 128, UU / 32, TT), blk, 0, stream>>>(ew1, w1tT, UU);

  // level 0
  gates0_kernel<<<BB / 8, 128, 0, stream>>>(x, gw0, gb0, gws, gbs, g0, gsArr);
  gemm_ple<0><<<g1z, blk, 0, stream>>>(xb, 0, w0sT, ebs0, nullptr, gsArr,
      sPart + 4 * BU, nullptr, esBf, nullptr, DD, DD);
  gemm_ple<1><<<g4z, blk, 0, stream>>>(xb, 0, w0tT, eb0, g0, gsArr,
      sPart, esBf, toBf, nullptr, DD, DD);
  share_cvt_kernel<<<(unsigned)(BU / 256), 256, 0, stream>>>(sPart, shareBf);

  // level 1
  gates1_kernel<<<dim3(BB / 8, TT), 128, 0, stream>>>(toBf, gw1, gb1, g1);
  gemm_ple<2><<<g1z, blk, 0, stream>>>(shareBf, 0, w1sT, ebs1, nullptr, nullptr,
      nullptr, nullptr, esBf, nullptr, UU, UU);
  gemm_ple<3><<<g4z, blk, 0, stream>>>(toBf, BU, w1tT, eb1, g1, nullptr,
      nullptr, esBf, nullptr, (float*)d_out, UU, UU);
}
